// GRAMTransformerLayer_34119220199536
// MI455X (gfx1250) — matrix operand order repaired and checked by reading, not compiled, not run on hardware
//
#include <hip/hip_runtime.h>
#include <hip/hip_bf16.h>

typedef _Float16 half_t;
typedef __attribute__((ext_vector_type(16))) _Float16 v16h;
typedef __attribute__((ext_vector_type(8)))  _Float16 v8h;
typedef __attribute__((ext_vector_type(8)))  float    v8f;

#define B_      4
#define T_Q     2048
#define TKV_    512
#define D_      1024
#define H_      16
#define HD_     64
#define HIDDEN_ 2736
#define EPS_    1e-6f

// ---------------------------------------------------------------- WMMA utils
__device__ __forceinline__ v8f wmma16(v16h a, v16h b, v8f c) {
  // (neg_a, A, neg_b, B, c_mod, C, reuse_a, reuse_b)
  return __builtin_amdgcn_wmma_f32_16x16x32_f16(false, a, false, b, (short)0, c,
                                                false, false);
}

__device__ __forceinline__ v16h cat8(v8h lo, v8h hi) {
  return __builtin_shufflevector(lo, hi, 0, 1, 2, 3, 4, 5, 6, 7,
                                 8, 9, 10, 11, 12, 13, 14, 15);
}

__device__ __forceinline__ v8f zero8() {
  v8f z = {0.f, 0.f, 0.f, 0.f, 0.f, 0.f, 0.f, 0.f};
  return z;
}

// A fragment (16x32 f16). base -> row 0 of a 16-row strip; `stride` in halves.
// ISA 7.12.2: lane = h*16+r holds row r, k = g*16 + h*8 + (e&7).
__device__ __forceinline__ v16h frag_a(const half_t* base, int stride, int lane) {
  const int h = lane >> 4, r = lane & 15;
  const half_t* p = base + (size_t)r * stride;
  v8h lo = *reinterpret_cast<const v8h*>(p + h * 8);
  v8h hi = *reinterpret_cast<const v8h*>(p + 16 + h * 8);
  return cat8(lo, hi);
}

// B fragment (32x16 f16) from Bt[n][k] storage (stride in halves).
// lane = h*16+c holds col c, k = g*16 + h*8 + (e&7).
__device__ __forceinline__ v16h frag_b(const half_t* base, int stride, int lane) {
  const int h = lane >> 4, c = lane & 15;
  const half_t* p = base + (size_t)c * stride + h * 8;
  v8h lo = *reinterpret_cast<const v8h*>(p);
  v8h hi = *reinterpret_cast<const v8h*>(p + 16);
  return cat8(lo, hi);
}

// CDNA5 async DMA: memory -> LDS, tracked by ASYNCcnt (no VGPR staging).
__device__ __forceinline__ void async_cp16(unsigned lds_off, const half_t* g) {
  asm volatile("global_load_async_to_lds_b128 %0, %1, off"
               :
               : "v"(lds_off), "v"(g)
               : "memory");
}
__device__ __forceinline__ void wait_async_le6() {
  asm volatile("s_wait_asynccnt 0x6" ::: "memory");
}
__device__ __forceinline__ void wait_async_0() {
  asm volatile("s_wait_asynccnt 0x0" ::: "memory");
}

// ---------------------------------------------------------------- GEMM
// C[M,N] (f32) = A[M,K] (f16 row-major) * Bt[N,K] (f16, pre-transposed weights)
// Workgroup tile 128x256; 8 waves as 2(M) x 4(N); wave tile 64x64 = 16 WMMA
// tiles -> 16 v_wmma per 32-wide K step per wave. ALIGNED path double-buffers
// LDS via GLOBAL_LOAD_ASYNC_TO_LDS_B128 and pipelines DMA against WMMA.
template <bool ALIGNED>
__global__ __launch_bounds__(256)
void gemm_f16f32(const half_t* __restrict__ A, const half_t* __restrict__ Bt,
                 float* __restrict__ C, int M, int N, int K) {
  __shared__ __align__(16) half_t sA[2][128 * 40];  // [row][k], stride 40 halves
  __shared__ __align__(16) half_t sB[2][256 * 40];  // [n][k]

  const int tid = threadIdx.x;
  const int lane = tid & 31, wid = tid >> 5;
  const int wm = wid >> 2, wn = wid & 3;  // 2 x 4 wave grid
  const int row0 = blockIdx.y * 128;
  const int col0 = blockIdx.x * 256;

  v8f acc[4][4];
#pragma unroll
  for (int i = 0; i < 4; ++i)
#pragma unroll
    for (int j = 0; j < 4; ++j) acc[i][j] = zero8();

  // 16-byte chunks: A = 512 chunks (2/thread), B = 1024 chunks (4/thread).
  auto issue_async = [&](int k0, int buf) {
#pragma unroll
    for (int i = 0; i < 2; ++i) {
      int c = tid + i * 256, rr = c >> 2, kk = (c & 3) * 8;
      async_cp16((unsigned)(size_t)&sA[buf][rr * 40 + kk],
                 A + (size_t)(row0 + rr) * K + k0 + kk);
    }
#pragma unroll
    for (int i = 0; i < 4; ++i) {
      int c = tid + i * 256, nr = c >> 2, kk = (c & 3) * 8;
      async_cp16((unsigned)(size_t)&sB[buf][nr * 40 + kk],
                 Bt + (size_t)(col0 + nr) * K + k0 + kk);
    }
  };

  auto stage_sync = [&](int k0, int buf) {
#pragma unroll
    for (int i = 0; i < 2; ++i) {
      int c = tid + i * 256, rr = c >> 2, kk = (c & 3) * 8;
      v8h vv = {};
      if (k0 + kk < K)  // K multiple of 16: 8-half chunk stays in bounds
        vv = *reinterpret_cast<const v8h*>(A + (size_t)(row0 + rr) * K + k0 + kk);
      *reinterpret_cast<v8h*>(&sA[buf][rr * 40 + kk]) = vv;
    }
#pragma unroll
    for (int i = 0; i < 4; ++i) {
      int c = tid + i * 256, nr = c >> 2, kk = (c & 3) * 8;
      v8h vv = {};
      if ((col0 + nr < N) && (k0 + kk < K))
        vv = *reinterpret_cast<const v8h*>(Bt + (size_t)(col0 + nr) * K + k0 + kk);
      *reinterpret_cast<v8h*>(&sB[buf][nr * 40 + kk]) = vv;
    }
  };

  auto compute = [&](int buf) {
    v16h af[4];
#pragma unroll
    for (int t = 0; t < 4; ++t)
      af[t] = frag_a(&sA[buf][(wm * 64 + t * 16) * 40], 40, lane);
#pragma unroll
    for (int u = 0; u < 4; ++u) {
      v16h bf = frag_b(&sB[buf][(wn * 64 + u * 16) * 40], 40, lane);
#pragma unroll
      for (int t = 0; t < 4; ++t) acc[t][u] = wmma16(af[t], bf, acc[t][u]);
    }
  };

  const int nk = (K + 31) / 32;
  if (ALIGNED) {
    issue_async(0, 0);
    for (int t = 0; t < nk; ++t) {
      const int buf = t & 1;
      if (t + 1 < nk) {
        issue_async((t + 1) * 32, buf ^ 1);  // overlap next DMA with this WMMA
        wait_async_le6();                    // tile t's 6 copies are done
      } else {
        wait_async_0();
      }
      __syncthreads();
      compute(buf);
      __syncthreads();
    }
  } else {
    for (int t = 0; t < nk; ++t) {
      stage_sync(t * 32, 0);
      __syncthreads();
      compute(0);
      __syncthreads();
    }
  }

  const int h = lane >> 4, cl = lane & 15;
#pragma unroll
  for (int t = 0; t < 4; ++t)
#pragma unroll
    for (int u = 0; u < 4; ++u)
#pragma unroll
      for (int i = 0; i < 8; ++i) {
        int r = row0 + wm * 64 + t * 16 + i + 8 * h;  // C layout: VGPR i -> i+8h
        int cc = col0 + wn * 64 + u * 16 + cl;
        if (ALIGNED || cc < N) C[(size_t)r * N + cc] = acc[t][u][i];
      }
}

// ---------------------------------------------------------------- Flash attn
// Qh/Kh/Vh: [B*H, T, 64] f16. Out: merged heads [B, Tq, D] f16.
// 4 waves/block; each wave owns a 16-row Q strip of a 64-row Q block.
// Scores stay in accumulators (never touch HBM); P relayouts via per-wave LDS.
template <bool CAUSAL>
__global__ __launch_bounds__(128)
void flash_attn(const half_t* __restrict__ Qh, const half_t* __restrict__ Kh,
                const half_t* __restrict__ Vh, half_t* __restrict__ Out,
                int Tq, int Tkv, float scale) {
  __shared__ __align__(16) half_t sVt[64 * 72];    // V^T tile: [d][t_local]
  __shared__ __align__(16) half_t sP[4][16 * 72];  // per-wave P strip [r][k]

  const int lane = threadIdx.x & 31, wid = threadIdx.x >> 5;
  const int hh = lane >> 4, lc = lane & 15;
  const int q0 = blockIdx.x * 64;
  const int bh = blockIdx.y;
  const int bb = bh / H_, head = bh % H_;

  const half_t* Qw = Qh + ((size_t)bh * Tq + q0 + wid * 16) * HD_;
  v16h qf[2];
#pragma unroll
  for (int ks = 0; ks < 2; ++ks) {
    const half_t* p = Qw + (size_t)lc * HD_ + ks * 32;
    v8h lo = *reinterpret_cast<const v8h*>(p + hh * 8);
    v8h hi = *reinterpret_cast<const v8h*>(p + 16 + hh * 8);
    qf[ks] = cat8(lo, hi);
  }

  const half_t* Kb = Kh + (size_t)bh * Tkv * HD_;
  const half_t* Vb = Vh + (size_t)bh * Tkv * HD_;

  float m[8], l[8];
  v8f o[4];
#pragma unroll
  for (int j = 0; j < 8; ++j) { m[j] = -1e30f; l[j] = 0.f; }
#pragma unroll
  for (int nd = 0; nd < 4; ++nd) o[nd] = zero8();

  const int kv_end = CAUSAL ? ((Tkv < q0 + 64) ? Tkv : (q0 + 64)) : Tkv;
  for (int kv0 = 0; kv0 < kv_end; kv0 += 64) {
    __syncthreads();
    // Stage V tile transposed: sVt[d][t] = V[kv0+t][d].
#pragma unroll
    for (int i = 0; i < 32; ++i) {
      int e = threadIdx.x + i * 128;
      int t = e >> 6, d = e & 63;
      sVt[d * 72 + t] = Vb[(size_t)(kv0 + t) * HD_ + d];
    }

    // S = Q*K^T; row-major K *is* Bt[n=t][k=d] for the K^T operand.
    v8f s[4];
#pragma unroll
    for (int n = 0; n < 4; ++n) {
      s[n] = zero8();
#pragma unroll
      for (int ks = 0; ks < 2; ++ks) {
        const half_t* kp =
            Kb + (size_t)(kv0 + n * 16 + lc) * HD_ + ks * 32 + hh * 8;
        v8h lo = *reinterpret_cast<const v8h*>(kp);
        v8h hi = *reinterpret_cast<const v8h*>(kp + 16);
        s[n] = wmma16(qf[ks], cat8(lo, hi), s[n]);
      }
    }

    float rowmax[8];
#pragma unroll
    for (int j = 0; j < 8; ++j) rowmax[j] = -1e30f;
#pragma unroll
    for (int n = 0; n < 4; ++n)
#pragma unroll
      for (int j = 0; j < 8; ++j) {
        float v = s[n][j] * scale;
        if (CAUSAL) {
          int t = kv0 + n * 16 + lc;
          int q = q0 + wid * 16 + j + 8 * hh;
          if (t > q) v = -1e30f;
        }
        s[n][j] = v;
        rowmax[j] = fmaxf(rowmax[j], v);
      }
#pragma unroll
    for (int j = 0; j < 8; ++j)
#pragma unroll
      for (int off = 1; off < 16; off <<= 1)
        rowmax[j] = fmaxf(rowmax[j], __shfl_xor(rowmax[j], off, 32));

    float rowsum[8];
#pragma unroll
    for (int j = 0; j < 8; ++j) {
      float mn = fmaxf(m[j], rowmax[j]);
      float alpha = __expf(m[j] - mn);
      m[j] = mn;
      l[j] *= alpha;
#pragma unroll
      for (int nd = 0; nd < 4; ++nd) o[nd][j] *= alpha;
      rowsum[j] = 0.f;
    }
#pragma unroll
    for (int n = 0; n < 4; ++n)
#pragma unroll
      for (int j = 0; j < 8; ++j) {
        float p = __expf(s[n][j] - m[j]);
        s[n][j] = p;
        rowsum[j] += p;
      }
#pragma unroll
    for (int j = 0; j < 8; ++j) {
#pragma unroll
      for (int off = 1; off < 16; off <<= 1)
        rowsum[j] += __shfl_xor(rowsum[j], off, 32);
      l[j] += rowsum[j];
    }

    // C-layout -> A-layout relayout of P via per-wave LDS strip.
    half_t* pw = sP[wid];
#pragma unroll
    for (int n = 0; n < 4; ++n)
#pragma unroll
      for (int j = 0; j < 8; ++j)
        pw[(j + 8 * hh) * 72 + n * 16 + lc] = (half_t)s[n][j];
    __syncthreads();  // sVt ready for everyone + P visible cross-lane

    // O += P * V
#pragma unroll
    for (int ks = 0; ks < 2; ++ks) {
      v16h pa = frag_a(pw + ks * 32, 72, lane);
#pragma unroll
      for (int nd = 0; nd < 4; ++nd) {
        const half_t* vp = &sVt[(nd * 16 + lc) * 72 + ks * 32 + hh * 8];
        v8h lo = *reinterpret_cast<const v8h*>(vp);
        v8h hi = *reinterpret_cast<const v8h*>(vp + 16);
        o[nd] = wmma16(pa, cat8(lo, hi), o[nd]);
      }
    }
  }

#pragma unroll
  for (int j = 0; j < 8; ++j) {
    float inv = (l[j] > 0.f) ? (1.f / l[j]) : 0.f;
    int q = q0 + wid * 16 + j + 8 * hh;
#pragma unroll
    for (int nd = 0; nd < 4; ++nd) {
      int d = nd * 16 + lc;
      Out[((size_t)bb * Tq + q) * D_ + head * HD_ + d] = (half_t)(o[nd][j] * inv);
    }
  }
}

// ---------------------------------------------------------------- elementwise
__global__ void f32_to_f16(const float* __restrict__ src,
                           half_t* __restrict__ dst, long n) {
  long i = (long)blockIdx.x * blockDim.x + threadIdx.x;
  if (i < n) dst[i] = (half_t)src[i];
}

// Weight convert + transpose: src[K][N] f32 -> dst[N][K] f16 (one-time cost).
__global__ void f32_to_f16_t(const float* __restrict__ src,
                             half_t* __restrict__ dst, int K, int N) {
  long i = (long)blockIdx.x * blockDim.x + threadIdx.x;
  if (i >= (long)K * N) return;
  int k = (int)(i / N), n = (int)(i % N);
  dst[(size_t)n * K + k] = (half_t)src[i];
}

// Split heads (+ optional RoPE) + f32->f16: [rows, D] -> [B, H, Trows, 64]
__global__ void split_rope(const float* __restrict__ src,
                           half_t* __restrict__ dst,
                           const float* __restrict__ freqs, int Trows,
                           int useRope, long npairs) {
  long i = (long)blockIdx.x * blockDim.x + threadIdx.x;
  if (i >= npairs) return;
  int dp = (int)(i & 511);  // D/2 = 512 pairs per row
  long bt = i >> 9;
  int t = (int)(bt % Trows);
  int b = (int)(bt / Trows);
  int head = dp >> 5, j = dp & 31;  // HD/2 = 32 pairs per head
  const float* p = src + bt * D_ + head * HD_ + 2 * j;
  float a = p[0], c = p[1];
  if (useRope) {
    float cs = freqs[((size_t)t * 32 + j) * 2];
    float sn = freqs[((size_t)t * 32 + j) * 2 + 1];
    float na = a * cs - c * sn;
    c = a * sn + c * cs;
    a = na;
  }
  size_t off = (((size_t)(b * H_ + head) * Trows) + t) * HD_ + 2 * j;
  dst[off] = (half_t)a;
  dst[off + 1] = (half_t)c;
}

// out = rmsnorm(a + badd) * g ; also emit f16 copy for the next GEMM.
__global__ __launch_bounds__(256)
void rmsnorm_add(const float* __restrict__ a, const float* __restrict__ badd,
                 const float* __restrict__ g, float* __restrict__ out,
                 half_t* __restrict__ out16) {
  const int row = blockIdx.x, tid = threadIdx.x;
  const size_t base = (size_t)row * D_;
  float v[4];
  float ss = 0.f;
#pragma unroll
  for (int i = 0; i < 4; ++i) {
    int c = tid + i * 256;
    v[i] = a[base + c] + badd[base + c];
    ss += v[i] * v[i];
  }
#pragma unroll
  for (int off = 16; off > 0; off >>= 1) ss += __shfl_xor(ss, off, 32);
  __shared__ float red[8];
  if ((tid & 31) == 0) red[tid >> 5] = ss;
  __syncthreads();
  float tot = 0.f;
#pragma unroll
  for (int w = 0; w < 8; ++w) tot += red[w];
  float r = rsqrtf(tot * (1.f / D_) + EPS_);
#pragma unroll
  for (int i = 0; i < 4; ++i) {
    int c = tid + i * 256;
    float ov = v[i] * r * g[c];
    out[base + c] = ov;
    if (out16) out16[base + c] = (half_t)ov;
  }
}

__global__ void silu_mul(const float* __restrict__ h1,
                         const float* __restrict__ h3,
                         half_t* __restrict__ outh, long n) {
  long i = (long)blockIdx.x * blockDim.x + threadIdx.x;
  if (i < n) {
    float x = h1[i];
    float s = x / (1.f + __expf(-x));
    outh[i] = (half_t)(s * h3[i]);
  }
}

// ---------------------------------------------------------------- launch
extern "C" void kernel_launch(void* const* d_in, const int* in_sizes, int n_in,
                              void* d_out, int out_size, void* d_ws,
                              size_t ws_size, hipStream_t stream) {
  (void)in_sizes; (void)n_in; (void)out_size; (void)ws_size;
  const float* x      = (const float*)d_in[0];
  const float* memory = (const float*)d_in[1];
  const float* freqs  = (const float*)d_in[2];
  // d_in[3] causal_mask: applied analytically inside flash_attn<true>
  const float* sa_wq = (const float*)d_in[4];
  const float* sa_wk = (const float*)d_in[5];
  const float* sa_wv = (const float*)d_in[6];
  const float* sa_wo = (const float*)d_in[7];
  const float* ca_wq = (const float*)d_in[8];
  const float* ca_wk = (const float*)d_in[9];
  const float* ca_wv = (const float*)d_in[10];
  const float* ca_wo = (const float*)d_in[11];
  const float* w1    = (const float*)d_in[12];  // dict order: w1, w3, w2
  const float* w3    = (const float*)d_in[13];
  const float* w2    = (const float*)d_in[14];
  const float* g_sa  = (const float*)d_in[15];
  const float* g_ca  = (const float*)d_in[16];
  const float* g_ffn = (const float*)d_in[17];
  float* out = (float*)d_out;

  const long NTOK = (long)B_ * T_Q;   // 8192
  const long NMEM = (long)B_ * TKV_;  // 2048

  char* ws = (char*)d_ws;
  size_t off = 0;
  auto alloc = [&](size_t bytes) {
    char* p = ws + off;
    off += (bytes + 255) & ~(size_t)255;
    return p;
  };
  half_t* X16 = (half_t*)alloc((size_t)NTOK * D_ * 2);
  half_t* M16 = (half_t*)alloc((size_t)NMEM * D_ * 2);
  char* Preg = alloc((size_t)3 * NTOK * D_ * 4);  // 100.7 MB
  float* P0 = (float*)Preg;
  float* P1 = (float*)(Preg + (size_t)NTOK * D_ * 4);
  float* P2 = (float*)(Preg + (size_t)2 * NTOK * D_ * 4);
  char* Hreg = alloc((size_t)4 * NTOK * D_ * 2 + (size_t)NTOK * D_ * 4);
  half_t* QH = (half_t*)Hreg;
  half_t* KH = (half_t*)(Hreg + (size_t)NTOK * D_ * 2);
  half_t* VH = (half_t*)(Hreg + (size_t)2 * NTOK * D_ * 2);
  half_t* AO = (half_t*)(Hreg + (size_t)3 * NTOK * D_ * 2);
  float*  G  = (float*)(Hreg + (size_t)4 * NTOK * D_ * 2);
  half_t* HH = (half_t*)alloc((size_t)NTOK * HIDDEN_ * 2);
  half_t* hSAq = (half_t*)alloc((size_t)D_ * D_ * 2);
  half_t* hSAk = (half_t*)alloc((size_t)D_ * D_ * 2);
  half_t* hSAv = (half_t*)alloc((size_t)D_ * D_ * 2);
  half_t* hSAo = (half_t*)alloc((size_t)D_ * D_ * 2);
  half_t* hCAq = (half_t*)alloc((size_t)D_ * D_ * 2);
  half_t* hCAk = (half_t*)alloc((size_t)D_ * D_ * 2);
  half_t* hCAv = (half_t*)alloc((size_t)D_ * D_ * 2);
  half_t* hCAo = (half_t*)alloc((size_t)D_ * D_ * 2);
  half_t* hW1 = (half_t*)alloc((size_t)D_ * HIDDEN_ * 2);
  half_t* hW3 = (half_t*)alloc((size_t)D_ * HIDDEN_ * 2);
  half_t* hW2 = (half_t*)alloc((size_t)HIDDEN_ * D_ * 2);
  // FFN aliases (dead buffers by then): H1 over P region, H3 over head region.
  float* H1 = (float*)Preg;
  float* H3 = (float*)Hreg;

  auto cvt = [&](const float* s, half_t* d, long n) {
    f32_to_f16<<<dim3((unsigned)((n + 255) / 256)), dim3(256), 0, stream>>>(s, d, n);
  };
  auto cvtT = [&](const float* s, half_t* d, int K, int N) {
    long n = (long)K * N;
    f32_to_f16_t<<<dim3((unsigned)((n + 255) / 256)), dim3(256), 0, stream>>>(s, d, K, N);
  };
  auto gemm = [&](const half_t* A, const half_t* Bm, float* C, int M, int N, int K) {
    dim3 grid((unsigned)((N + 255) / 256), (unsigned)((M + 127) / 128));
    if ((N % 256 == 0) && (K % 32 == 0))
      gemm_f16f32<true><<<grid, dim3(256), 0, stream>>>(A, Bm, C, M, N, K);
    else
      gemm_f16f32<false><<<grid, dim3(256), 0, stream>>>(A, Bm, C, M, N, K);
  };

  // --- precision conversion (activations) + transposed weight conversion
  cvt(x, X16, NTOK * D_);
  cvt(memory, M16, NMEM * D_);
  cvtT(sa_wq, hSAq, D_, D_);
  cvtT(sa_wk, hSAk, D_, D_);
  cvtT(sa_wv, hSAv, D_, D_);
  cvtT(sa_wo, hSAo, D_, D_);
  cvtT(ca_wq, hCAq, D_, D_);
  cvtT(ca_wk, hCAk, D_, D_);
  cvtT(ca_wv, hCAv, D_, D_);
  cvtT(ca_wo, hCAo, D_, D_);
  cvtT(w1, hW1, D_, HIDDEN_);
  cvtT(w3, hW3, D_, HIDDEN_);
  cvtT(w2, hW2, HIDDEN_, D_);

  const float scale = 0.125f;  // HD^-0.5
  const long npq = NTOK * (D_ / 2);
  const long npm = NMEM * (D_ / 2);

  // --- self-attention
  gemm(X16, hSAq, P0, (int)NTOK, D_, D_);
  gemm(X16, hSAk, P1, (int)NTOK, D_, D_);
  gemm(X16, hSAv, P2, (int)NTOK, D_, D_);
  split_rope<<<dim3((unsigned)((npq + 255) / 256)), dim3(256), 0, stream>>>(
      P0, QH, freqs, T_Q, 1, npq);
  split_rope<<<dim3((unsigned)((npq + 255) / 256)), dim3(256), 0, stream>>>(
      P1, KH, freqs, T_Q, 1, npq);
  split_rope<<<dim3((unsigned)((npq + 255) / 256)), dim3(256), 0, stream>>>(
      P2, VH, freqs, T_Q, 0, npq);
  flash_attn<true><<<dim3(T_Q / 64, B_ * H_), dim3(128), 0, stream>>>(
      QH, KH, VH, AO, T_Q, T_Q, scale);
  gemm(AO, hSAo, G, (int)NTOK, D_, D_);
  rmsnorm_add<<<dim3((unsigned)NTOK), dim3(256), 0, stream>>>(x, G, g_sa, out, X16);

  // --- cross-attention (no rope, no mask)
  gemm(X16, hCAq, P0, (int)NTOK, D_, D_);
  gemm(M16, hCAk, P1, (int)NMEM, D_, D_);
  gemm(M16, hCAv, P2, (int)NMEM, D_, D_);
  split_rope<<<dim3((unsigned)((npq + 255) / 256)), dim3(256), 0, stream>>>(
      P0, QH, freqs, T_Q, 0, npq);
  split_rope<<<dim3((unsigned)((npm + 255) / 256)), dim3(256), 0, stream>>>(
      P1, KH, freqs, TKV_, 0, npm);
  split_rope<<<dim3((unsigned)((npm + 255) / 256)), dim3(256), 0, stream>>>(
      P2, VH, freqs, TKV_, 0, npm);
  flash_attn<false><<<dim3(T_Q / 64, B_ * H_), dim3(128), 0, stream>>>(
      QH, KH, VH, AO, T_Q, TKV_, scale);
  gemm(AO, hCAo, G, (int)NTOK, D_, D_);
  rmsnorm_add<<<dim3((unsigned)NTOK), dim3(256), 0, stream>>>(out, G, g_ca, out, X16);

  // --- SwiGLU FFN
  gemm(X16, hW1, H1, (int)NTOK, HIDDEN_, D_);
  gemm(X16, hW3, H3, (int)NTOK, HIDDEN_, D_);
  const long nh = NTOK * (long)HIDDEN_;
  silu_mul<<<dim3((unsigned)((nh + 255) / 256)), dim3(256), 0, stream>>>(H1, H3, HH, nh);
  gemm(HH, hW2, G, (int)NTOK, D_, HIDDEN_);
  rmsnorm_add<<<dim3((unsigned)NTOK), dim3(256), 0, stream>>>(out, G, g_ffn, out,
                                                             (half_t*)nullptr);
}
